// VQVAE_2181843386753
// MI455X (gfx1250) — matrix operand order repaired and checked by reading, not compiled, not run on hardware
//
#include <hip/hip_runtime.h>

// ---------------------------------------------------------------------------
// MI455X (gfx1250) VQ-VAE forward. wave32; WMMA 16x16x32 bf16 for all convs
// (implicit GEMM, weights staged in LDS), WMMA 16x16x4 f32 for the VQ
// distance GEMM. Forward-pass algebra: ste == quantized, so the decoder reads
// the quantized codes; e_q_loss = 1.25 * mean((q - z)^2); x_recon is never
// materialized (MSE folded into dec3 epilogue).
// Async global->LDS staging (ASYNCcnt path) for codebook and raw weights via
// __builtin_amdgcn_global_load_async_to_lds_b128 (probe: param0 is
// 'int4 addrspace(1)*'), plus global_prefetch_b8 one K-step ahead.
// Workspace layout (liveness-aliased, needs ws_size >= ~412 MB):
//   [0 .. 268MB)   : a1 (bf16, 67MB) -> z (f32 @ +128MB, 16.8MB) -> y2 (bf16)
//   [268 .. 402MB) : a2 (bf16, 33.5MB) -> y1 (bf16, 134MB)
//   [402 .. 411MB) : e (bf16 NHWC, 8.4MB)
//   [411MB ..)     : float acc[2]  (vq_sum, mse_sum)
// ---------------------------------------------------------------------------

typedef __attribute__((ext_vector_type(16))) __bf16       v16bf;
typedef __attribute__((ext_vector_type(8)))  float        v8f;
typedef __attribute__((ext_vector_type(2)))  float        v2f;
typedef __attribute__((ext_vector_type(4)))  unsigned int v4u;
typedef int i128v __attribute__((vector_size(16)));   // int4, matches builtin

#if defined(__has_builtin)
#if __has_builtin(__builtin_amdgcn_global_load_async_to_lds_b128) && \
    __has_builtin(__builtin_amdgcn_s_wait_asynccnt)
#define HAVE_ASYNC_LDS 1
#endif
#endif
#ifndef HAVE_ASYNC_LDS
#define HAVE_ASYNC_LDS 0
#endif

#define AS1Q __attribute__((address_space(1)))
#define AS3Q __attribute__((address_space(3)))

#if HAVE_ASYNC_LDS
// Async DMA of 16 bytes global -> LDS (tracked by ASYNCcnt, no VGPR data).
static __device__ __forceinline__ void g2l_b128(const void* g, void* l) {
  __builtin_amdgcn_global_load_async_to_lds_b128(
      (AS1Q i128v*)g, (AS3Q i128v*)l, 0, 0);
}
static __device__ __forceinline__ void async_wait0() {
  __builtin_amdgcn_s_wait_asynccnt(0);
}
#endif

static __device__ __forceinline__ unsigned short f2bf(float f) {
  unsigned u = __builtin_bit_cast(unsigned, f);
  u += 0x7fffu + ((u >> 16) & 1u);          // round-to-nearest-even
  return (unsigned short)(u >> 16);
}

static __device__ __forceinline__ float wave_sum(float v) {
#pragma unroll
  for (int m = 16; m >= 1; m >>= 1) v += __shfl_xor(v, m, 32);
  return v;
}

union AFrag { unsigned short u[16]; v16bf v; };
union BFrag { v4u q[2];             v16bf v; };

static __device__ __forceinline__ v8f wmma_bf16(v16bf a, v16bf b, v8f c) {
  return __builtin_amdgcn_wmma_f32_16x16x32_bf16(false, a, false, b,
                                                 (short)0, c, false, false);
}
static __device__ __forceinline__ v8f wmma_f32k4(v2f a, v2f b, v8f c) {
  return __builtin_amdgcn_wmma_f32_16x16x4_f32(false, a, false, b,
                                               (short)0, c, false, false);
}

// ---------------------------------------------------------------------------
// Generic implicit-GEMM conv / transposed-conv.
//  M = 16 consecutive output pixels per wave (one row; WOUT % 16 == 0),
//  N = output channels (16-wide tiles), K = CIN*KH*KW padded to mult of 32.
//  SRCK: 0 = bf16 NCHW, 1 = f32 NCHW (network input), 2 = bf16 NHWC (VQ codes)
//  EPI : 0 = relu -> bf16 NCHW store, 1 = f32 NCHW store (latents z),
//        2 = fused MSE vs xref (atomicAdd into acc[1], no store)
// ---------------------------------------------------------------------------
template <int CIN, int COUT, int KH, int KW, int HIN, int WIN, int HOUT,
          int WOUT, int STRIDE, int PAD, bool TRANSP, int SRCK, int EPI>
__global__ __launch_bounds__(256) void conv_wmma(
    const void* __restrict__ src, const float* __restrict__ wgt,
    const float* __restrict__ bias, void* __restrict__ dst,
    const float* __restrict__ xref, float* __restrict__ accum) {
  constexpr int KHW   = KH * KW;
  constexpr int KK    = CIN * KHW;
  constexpr int KPAD  = (KK + 31) & ~31;
  constexpr int NK    = KPAD / 32;
  constexpr int COUTP = (COUT + 15) & ~15;
  constexpr int NT    = COUTP / 16;
  constexpr int WTOT  = COUT * KK;          // raw f32 weight elements

  __shared__ __align__(16) unsigned short wlds[COUTP * KPAD];
#if HAVE_ASYNC_LDS
  // Stage raw f32 weights with the async-DMA path, then convert from LDS.
  __shared__ __align__(16) float wraw[WTOT];
  for (int off = threadIdx.x * 16; off < WTOT * 4; off += 256 * 16)
    g2l_b128((const char*)wgt + off, (char*)wraw + off);
  async_wait0();
  __syncthreads();
  const float* wsrc = wraw;
#else
  const float* wsrc = wgt;
#endif

  // Stage weights to LDS as bf16, [co][k] so B fragments are 2x ds_load_b128.
  // Transposed convs use w[ci][co][ky][kx] directly (gather form needs no flip).
  for (int i = threadIdx.x; i < COUTP * KPAD; i += 256) {
    int co = i / KPAD, k = i % KPAD;
    unsigned short hv = 0;
    if (co < COUT && k < KK) {
      int ci = k / KHW, rem = k % KHW, ky = rem / KW, kx = rem % KW;
      float f = TRANSP ? wsrc[((ci * COUT + co) * KH + ky) * KW + kx]
                       : wsrc[((co * CIN + ci) * KH + ky) * KW + kx];
      hv = f2bf(f);
    }
    wlds[i] = hv;
  }
  __syncthreads();

  const int lane = threadIdx.x & 31;
  const int wid  = threadIdx.x >> 5;
  const int lh   = lane >> 4;        // lane-half (K split in A/B layouts)
  const int ml   = lane & 15;        // A row / B,C column within tile
  const int img  = blockIdx.y;
  const int m0   = (blockIdx.x * 8 + wid) * 16;   // 16 pixels, same output row
  const int oy   = m0 / WOUT;
  const int ox0  = m0 % WOUT;
  const int ox   = ox0 + ml;

  // Per-k gather address (nullptr if out of bounds / padding tap).
  auto gaddr = [&](int k) -> const void* {
    int ci = k / KHW, rem = k % KHW, ky = rem / KW, kx = rem % KW;
    int iy, ix;
    bool ok;
    if (TRANSP) {
      int ty = oy + PAD - ky, tx = ox + PAD - kx;
      iy = ty / STRIDE; ix = tx / STRIDE;
      ok = (ty >= 0) && (tx >= 0) && (ty % STRIDE == 0) &&
           (tx % STRIDE == 0) && (iy < HIN) && (ix < WIN);
    } else {
      iy = oy * STRIDE - PAD + ky; ix = ox * STRIDE - PAD + kx;
      ok = (iy >= 0) && (iy < HIN) && (ix >= 0) && (ix < WIN);
    }
    if (!ok) return (const void*)nullptr;
    if constexpr (SRCK == 1)
      return (const void*)((const float*)src +
                           ((img * CIN + ci) * HIN + iy) * WIN + ix);
    else if constexpr (SRCK == 2)
      return (const void*)((const unsigned short*)src +
                           ((img * HIN + iy) * WIN + ix) * CIN + ci);
    else
      return (const void*)((const unsigned short*)src +
                           ((img * CIN + ci) * HIN + iy) * WIN + ix);
  };

  v8f accv[NT];
  const v8f vzero = {0.f, 0.f, 0.f, 0.f, 0.f, 0.f, 0.f, 0.f};
#pragma unroll
  for (int t = 0; t < NT; ++t) accv[t] = vzero;

  for (int q = 0; q < NK; ++q) {
    // Prefetch next K-chunk's row for this lane (global_prefetch_b8).
    if (q + 1 < NK) {
      int kp = (q + 1) * 32 + (lh << 3);
      if (kp < KK) {
        const void* pp = gaddr(kp);
        if (pp) __builtin_prefetch(pp, 0, 1);
      }
    }
    // A fragment: 16-bit A 16x32 layout -> K = lh*16 + j (contiguous, matches B).
    AFrag af;
#pragma unroll
    for (int j = 0; j < 16; ++j) {
      int k = q * 32 + (lh << 4) + j;
      unsigned short hv = 0;
      if (k < KK) {
        const void* p = gaddr(k);
        if (p) {
          if constexpr (SRCK == 1) hv = f2bf(*(const float*)p);
          else                     hv = *(const unsigned short*)p;
        }
      }
      af.u[j] = hv;
    }
#pragma unroll
    for (int t = 0; t < NT; ++t) {
      // Dense 16-bit B 32x16: lane = column, K = q*32 + lh*16 + j (contiguous).
      BFrag bf;
      const v4u* bp =
          (const v4u*)&wlds[(t * 16 + ml) * KPAD + q * 32 + (lh << 4)];
      bf.q[0] = bp[0];
      bf.q[1] = bp[1];
      accv[t] = wmma_bf16(af.v, bf.v, accv[t]);
    }
  }

  // C/D layout: lane column n = ml, rows M = r + 8*lh (r = vgpr index).
  if constexpr (EPI == 0) {  // relu -> bf16 NCHW
#pragma unroll
    for (int t = 0; t < NT; ++t) {
      int co = t * 16 + ml;
      float bco = bias[co];
      unsigned short* dp = (unsigned short*)dst +
          ((img * COUT + co) * HOUT + oy) * WOUT + ox0 + (lh << 3);
#pragma unroll
      for (int i = 0; i < 4; ++i) {
        float v0 = fmaxf(accv[t][2 * i] + bco, 0.f);
        float v1 = fmaxf(accv[t][2 * i + 1] + bco, 0.f);
        unsigned pw = (unsigned)f2bf(v0) | ((unsigned)f2bf(v1) << 16);
        *(unsigned*)(dp + 2 * i) = pw;
      }
    }
  } else if constexpr (EPI == 1) {  // f32 NCHW (latents z)
#pragma unroll
    for (int t = 0; t < NT; ++t) {
      int co = t * 16 + ml;
      float bco = bias[co];
      float* fp = (float*)dst +
          ((img * COUT + co) * HOUT + oy) * WOUT + ox0 + (lh << 3);
#pragma unroll
      for (int i = 0; i < 4; ++i) {
        v2f pw = {accv[t][2 * i] + bco, accv[t][2 * i + 1] + bco};
        *(v2f*)(fp + 2 * i) = pw;
      }
    }
  } else {  // EPI == 2: fused sum((out - xref)^2), COUT <= 16 (dec3: 3)
    int co = ml;
    float s = 0.f;
    if (co < COUT) {
      float bco = bias[co];
      const float* xp = xref +
          ((img * COUT + co) * HOUT + oy) * WOUT + ox0 + (lh << 3);
#pragma unroll
      for (int r = 0; r < 8; ++r) {
        float d = (accv[0][r] + bco) - xp[r];
        s += d * d;
      }
    }
    s = wave_sum(s);
    if (lane == 0) atomicAdd(accum + 1, s);
  }
}

// ---------------------------------------------------------------------------
// Vector quantization: dist = ||z||^2 + ||c||^2 - 2 z.c via f32 WMMA 16x16x4,
// K=16 (4 chained steps), 32 code tiles of 16. argmin via 16-lane shfl_xor
// min-reduction carrying (dist, idx). Emits quantized rows (bf16 NHWC) and
// accumulates sum((q - z)^2). Codebook staged via async DMA when available.
// ---------------------------------------------------------------------------
__global__ __launch_bounds__(256) void vq_kernel(const float* __restrict__ z,
                                                 const float* __restrict__ cb,
                                                 unsigned short* __restrict__ e,
                                                 float* __restrict__ accum) {
  __shared__ __align__(16) float cbl[512 * 16];
  __shared__ float ccn[512];
  __shared__ float zzl[8][16];

#if HAVE_ASYNC_LDS
  for (int off = threadIdx.x * 16; off < 512 * 16 * 4; off += 256 * 16)
    g2l_b128((const char*)cb + off, (char*)cbl + off);
  async_wait0();
  __syncthreads();
#else
  for (int i = threadIdx.x; i < 512 * 16; i += 256) cbl[i] = cb[i];
  __syncthreads();
#endif
  for (int i = threadIdx.x; i < 512; i += 256) {
    float s = 0.f;
#pragma unroll
    for (int k = 0; k < 16; ++k) { float c = cbl[i * 16 + k]; s += c * c; }
    ccn[i] = s;
  }
  __syncthreads();

  const int lane = threadIdx.x & 31;
  const int wid  = threadIdx.x >> 5;
  const int lh   = lane >> 4;
  const int ml   = lane & 15;
  const int row0 = (blockIdx.x * 8 + wid) * 16;   // 16 latent positions
  const int img  = row0 >> 12;                    // 4096 positions / image
  const int pos0 = row0 & 4095;

  // A fragments (f32 A 16x4: vgpr0=K(2lh), vgpr1=K(2lh+1)) + row norms.
  v2f a[4];
  float zzp = 0.f;
#pragma unroll
  for (int q = 0; q < 4; ++q) {
    int k0 = 4 * q + 2 * lh;
    float x0 = z[(img * 16 + k0) * 4096 + pos0 + ml];
    float x1 = z[(img * 16 + k0 + 1) * 4096 + pos0 + ml];
    a[q].x = x0; a[q].y = x1;
    zzp += x0 * x0 + x1 * x1;
  }
  zzp += __shfl_xor(zzp, 16, 32);          // combine the two K halves
  if (lane < 16) zzl[wid][lane] = zzp;     // norm of row `lane`
  float zzr[8];
#pragma unroll
  for (int r = 0; r < 8; ++r) zzr[r] = zzl[wid][r + (lh << 3)];

  float best[8]; int bidx[8];
#pragma unroll
  for (int r = 0; r < 8; ++r) { best[r] = 3.4e38f; bidx[r] = 0; }

  const v8f vzero = {0.f, 0.f, 0.f, 0.f, 0.f, 0.f, 0.f, 0.f};
  for (int t = 0; t < 32; ++t) {
    int code = t * 16 + ml;
    v2f b[4];
#pragma unroll
    for (int q = 0; q < 4; ++q) {
      int k0 = 4 * q + 2 * lh;
      b[q].x = cbl[code * 16 + k0];
      b[q].y = cbl[code * 16 + k0 + 1];
    }
    v8f d = vzero;
#pragma unroll
    for (int q = 0; q < 4; ++q) d = wmma_f32k4(a[q], b[q], d);
    float cc = ccn[code];
#pragma unroll
    for (int r = 0; r < 8; ++r) {
      float dist = zzr[r] + cc - 2.0f * d[r];
      if (dist < best[r]) { best[r] = dist; bidx[r] = code; }
    }
  }
  // min across the 16 lanes holding the same rows
#pragma unroll
  for (int m = 1; m < 16; m <<= 1) {
#pragma unroll
    for (int r = 0; r < 8; ++r) {
      float od = __shfl_xor(best[r], m, 32);
      int   oi = __shfl_xor(bidx[r], m, 32);
      if (od < best[r]) { best[r] = od; bidx[r] = oi; }
    }
  }
  // lanes ml<8 each own one row: row = row0 + (lane&7) + 8*lh
  int rsel = lane & 7;
  int idx = bidx[0];
#pragma unroll
  for (int r = 1; r < 8; ++r) if (rsel == r) idx = bidx[r];
  float sq = 0.f;
  if (ml < 8) {
    int m = rsel + (lh << 3);
    int row = row0 + m;
    union { unsigned short h[16]; v4u q[2]; } pk;
#pragma unroll
    for (int k = 0; k < 16; ++k) {
      float cv = cbl[idx * 16 + k];
      float zv = z[(img * 16 + k) * 4096 + pos0 + m];
      float dd = cv - zv;
      sq += dd * dd;
      pk.h[k] = f2bf(cv);
    }
    v4u* ep = (v4u*)(e + row * 16);        // NHWC row, 32B aligned
    ep[0] = pk.q[0];
    ep[1] = pk.q[1];
  }
  sq = wave_sum(sq);
  if (lane == 0) atomicAdd(accum + 0, sq);
}

__global__ void zero_acc(float* accum) {
  if (threadIdx.x < 2) accum[threadIdx.x] = 0.f;
}

__global__ void finalize_k(const float* __restrict__ accum,
                           float* __restrict__ out) {
  float vq_mean = accum[0] * (1.0f / 4194304.0f);    // 262144 rows * 16
  float mse     = accum[1] * (1.0f / 12582912.0f);   // 64*3*256*256
  out[0] = 1.25f * vq_mean;   // q_latent + 0.25*e_latent (equal in fwd)
  out[1] = mse;               // DATA_VARIANCE == 1.0
  out[2] = mse;
}

extern "C" void kernel_launch(void* const* d_in, const int* in_sizes, int n_in,
                              void* d_out, int out_size, void* d_ws,
                              size_t ws_size, hipStream_t stream) {
  (void)in_sizes; (void)n_in; (void)out_size; (void)ws_size;
  const float* x   = (const float*)d_in[0];
  const float* ew1 = (const float*)d_in[1];
  const float* eb1 = (const float*)d_in[2];
  const float* ew2 = (const float*)d_in[3];
  const float* eb2 = (const float*)d_in[4];
  const float* ew3 = (const float*)d_in[5];
  const float* eb3 = (const float*)d_in[6];
  const float* cb  = (const float*)d_in[7];
  const float* dw1 = (const float*)d_in[8];
  const float* db1 = (const float*)d_in[9];
  const float* dw2 = (const float*)d_in[10];
  const float* db2 = (const float*)d_in[11];
  const float* dw3 = (const float*)d_in[12];
  const float* db3 = (const float*)d_in[13];

  char* ws = (char*)d_ws;
  unsigned short* a1  = (unsigned short*)(ws + 0);                    // 67MB
  float*          zb  = (float*)(ws + (size_t)134217728);             // 16.8MB
  unsigned short* y2b = (unsigned short*)(ws + 0);                    // 268MB
  unsigned short* a2  = (unsigned short*)(ws + (size_t)268435456);    // 33.5MB
  unsigned short* y1b = (unsigned short*)(ws + (size_t)268435456);    // 134MB
  unsigned short* eb16= (unsigned short*)(ws + (size_t)402653184);    // 8.4MB
  float*          acc = (float*)(ws + (size_t)411041792);             // 2 f32

  dim3 B(256);
  zero_acc<<<dim3(1), dim3(64), 0, stream>>>(acc);
  // Encoder
  conv_wmma<3, 32, 3, 3, 256, 256, 128, 128, 2, 1, false, 1, 0>
      <<<dim3(128, 64), B, 0, stream>>>(x, ew1, eb1, a1, nullptr, nullptr);
  conv_wmma<32, 64, 3, 3, 128, 128, 64, 64, 2, 1, false, 0, 0>
      <<<dim3(32, 64), B, 0, stream>>>(a1, ew2, eb2, a2, nullptr, nullptr);
  conv_wmma<64, 16, 1, 1, 64, 64, 64, 64, 1, 0, false, 0, 1>
      <<<dim3(32, 64), B, 0, stream>>>(a2, ew3, eb3, zb, nullptr, nullptr);
  // Vector quantization
  vq_kernel<<<dim3(2048), B, 0, stream>>>(zb, cb, eb16, acc);
  // Decoder (transposed convs in gather form)
  conv_wmma<16, 64, 3, 3, 64, 64, 128, 128, 2, 1, true, 2, 0>
      <<<dim3(128, 64), B, 0, stream>>>(eb16, dw1, db1, y1b, nullptr, nullptr);
  conv_wmma<64, 32, 3, 3, 128, 128, 256, 256, 2, 1, true, 0, 0>
      <<<dim3(512, 64), B, 0, stream>>>(y1b, dw2, db2, y2b, nullptr, nullptr);
  conv_wmma<32, 3, 3, 3, 256, 256, 256, 256, 1, 1, true, 0, 2>
      <<<dim3(512, 64), B, 0, stream>>>(y2b, dw3, db3, nullptr, x, acc);
  finalize_k<<<dim3(1), dim3(1), 0, stream>>>(acc, (float*)d_out);
}
